// Tokenizer_9294309229076
// MI455X (gfx1250) — matrix operand order repaired and checked by reading, not compiled, not run on hardware
//
#include <hip/hip_runtime.h>
#include <hip/hip_bf16.h>
#include <stdint.h>

// Problem constants (match reference)
#define B_  16
#define T_  1024
#define C_  512
#define V_  2048
#define N_  (B_ * T_)
#define THETA_THRESH 0.8f

// LDS A-tile row stride in bf16 elements: 512 + 8 pad -> 1040B/row = 4-bank
// shift per row so 16 lanes doing b128 fragment reads hit all 64 banks.
#define LDSTR 520

typedef __attribute__((ext_vector_type(16))) __bf16 v16bf;
typedef __attribute__((ext_vector_type(8)))  float  v8f;

union Frag16 { v16bf v; uint4 u[2]; };

__device__ __forceinline__ unsigned short f2bf_rne(float f) {
  unsigned u = __float_as_uint(f);
  u += 0x7FFFu + ((u >> 16) & 1u);          // round-to-nearest-even
  return (unsigned short)(u >> 16);
}
__device__ __forceinline__ float bf2f(unsigned short h) {
  return __uint_as_float(((unsigned)h) << 16);
}

// ---------------------------------------------------------------------------
// Zero the loss accumulators (ws is only poisoned once by the harness).
// ---------------------------------------------------------------------------
__global__ void zero_accum(float* accum) {
  if (threadIdx.x < 8) accum[threadIdx.x] = 0.0f;
}

// ---------------------------------------------------------------------------
// Split codebook into bf16 hi/lo planes and precompute 0.5*||e||^2 per code.
// Row 0 is the pad row: give it +huge "distance bias" so it is never argmin.
// ---------------------------------------------------------------------------
__global__ void __launch_bounds__(128)
prep_codebook(const float* __restrict__ cb,
              unsigned short* __restrict__ cb_hi,
              unsigned short* __restrict__ cb_lo,
              float* __restrict__ e2half) {
  __shared__ float red[128];
  const int v = blockIdx.x, tid = threadIdx.x;
  const float* row = cb + (size_t)v * C_;
  float s = 0.0f;
  for (int c = tid; c < C_; c += 128) {
    float f = row[c];
    unsigned short h = f2bf_rne(f);
    float lo = f - bf2f(h);
    cb_hi[(size_t)v * C_ + c] = h;
    cb_lo[(size_t)v * C_ + c] = f2bf_rne(lo);
    s += f * f;
  }
  red[tid] = s; __syncthreads();
  for (int o = 64; o > 0; o >>= 1) {
    if (tid < o) red[tid] += red[tid + o];
    __syncthreads();
  }
  if (tid == 0) e2half[v] = (v == 0) ? 3.0e38f : 0.5f * red[0];
}

// ---------------------------------------------------------------------------
// Nearest-code search via bf16x3-split WMMA GEMM + running argmin.
// One wave (32 threads) per 16-row M-tile; sweeps 64 pairs of 16-col tiles.
// Per K-chunk: 6 WMMAs across 4 independent accumulator chains (2 col tiles
// x {hi*hi, hi*lo+lo*hi}) so issue is not bound by the D->C dependency.
// metric = 0.5*||e||^2 - z.e   (||z||^2 dropped: constant per row)
// ---------------------------------------------------------------------------
__global__ void __launch_bounds__(32)
argmin_wmma(const float* __restrict__ z,
            const unsigned short* __restrict__ cb_hi,
            const unsigned short* __restrict__ cb_lo,
            const float* __restrict__ e2half,
            int* __restrict__ out_idx) {
  __shared__ unsigned short Ahi[16 * LDSTR];
  __shared__ unsigned short Alo[16 * LDSTR];

  const int lane    = threadIdx.x;          // 0..31
  const int rowBase = blockIdx.x * 16;

  // Stage A tile (16 rows x 512 K) into LDS as bf16 hi/lo.
  {
    const int r  = lane >> 1;
    const int c0 = (lane & 1) * 256;
    const float* src = z + (size_t)(rowBase + r) * C_ + c0;
    unsigned short* dh = Ahi + r * LDSTR + c0;
    unsigned short* dl = Alo + r * LDSTR + c0;
    #pragma unroll 4
    for (int c = 0; c < 256; c += 4) {
      float4 f = *(const float4*)(src + c);
      unsigned short hx = f2bf_rne(f.x), hy = f2bf_rne(f.y);
      unsigned short hz = f2bf_rne(f.z), hw = f2bf_rne(f.w);
      uint2 ph, pl;
      ph.x = (unsigned)hx | ((unsigned)hy << 16);
      ph.y = (unsigned)hz | ((unsigned)hw << 16);
      pl.x = (unsigned)f2bf_rne(f.x - bf2f(hx)) |
             ((unsigned)f2bf_rne(f.y - bf2f(hy)) << 16);
      pl.y = (unsigned)f2bf_rne(f.z - bf2f(hz)) |
             ((unsigned)f2bf_rne(f.w - bf2f(hw)) << 16);
      *(uint2*)(dh + c) = ph;
      *(uint2*)(dl + c) = pl;
    }
  }
  __syncthreads();

  const int m = lane & 15;     // A row (within tile) for this lane / C column n
  const int h = lane >> 4;     // lane half

  float minv[8]; int mini[8];
  #pragma unroll
  for (int r = 0; r < 8; ++r) { minv[r] = 3.4e38f; mini[r] = 0; }

  for (int ct = 0; ct < V_ / 16; ct += 2) {
    const int colA = ct * 16 + m;            // codebook row = B/C column, tile A
    const int colB = colA + 16;              // tile B
    const unsigned short* bhA = cb_hi + (size_t)colA * C_;
    const unsigned short* blA = cb_lo + (size_t)colA * C_;
    const unsigned short* bhB = cb_hi + (size_t)colB * C_;
    const unsigned short* blB = cb_lo + (size_t)colB * C_;
    if (ct + 2 < V_ / 16) {                  // gfx1250 global_prefetch_b8
      __builtin_prefetch(bhA + (size_t)32 * C_, 0, 1);
      __builtin_prefetch(blA + (size_t)32 * C_, 0, 1);
      __builtin_prefetch(bhB + (size_t)32 * C_, 0, 1);
      __builtin_prefetch(blB + (size_t)32 * C_, 0, 1);
    }

    v8f accPA = {}, accQA = {};              // tile A: hi*hi / cross terms
    v8f accPB = {}, accQB = {};              // tile B: hi*hi / cross terms
    #pragma unroll 2
    for (int k0 = 0; k0 < C_; k0 += 32) {
      Frag16 ah, al, bha, bla, bhb, blb;
      // A fragment (contiguous K order, not AMD's table order): lane half h holds K chunk
      // [16h,16h+16) of row m (matches B's K order).
      const int aoff = m * LDSTR + k0 + 16 * h;
      ah.u[0] = *(const uint4*)(Ahi + aoff);
      ah.u[1] = *(const uint4*)(Ahi + aoff + 8);
      al.u[0] = *(const uint4*)(Alo + aoff);
      al.u[1] = *(const uint4*)(Alo + aoff + 8);
      // B fragment (same contiguous K order as A): lane half h holds K=[16h,16h+16)
      // of column n = m, i.e. codebook row `col`, contiguous 32B.
      const int boff = k0 + 16 * h;
      bha.u[0] = *(const uint4*)(bhA + boff);
      bha.u[1] = *(const uint4*)(bhA + boff + 8);
      bla.u[0] = *(const uint4*)(blA + boff);
      bla.u[1] = *(const uint4*)(blA + boff + 8);
      bhb.u[0] = *(const uint4*)(bhB + boff);
      bhb.u[1] = *(const uint4*)(bhB + boff + 8);
      blb.u[0] = *(const uint4*)(blB + boff);
      blb.u[1] = *(const uint4*)(blB + boff + 8);
      // bf16x3 split per tile: hi*hi + hi*lo + lo*hi ~= fp32 dot.
      // 4 independent chains (accPA/accQA/accPB/accQB) for WMMA pipe ILP.
      accPA = __builtin_amdgcn_wmma_f32_16x16x32_bf16(false, ah.v, false, bha.v,
                                                      (short)0, accPA, false, false);
      accPB = __builtin_amdgcn_wmma_f32_16x16x32_bf16(false, ah.v, false, bhb.v,
                                                      (short)0, accPB, false, false);
      accQA = __builtin_amdgcn_wmma_f32_16x16x32_bf16(false, ah.v, false, bla.v,
                                                      (short)0, accQA, false, false);
      accQB = __builtin_amdgcn_wmma_f32_16x16x32_bf16(false, ah.v, false, blb.v,
                                                      (short)0, accQB, false, false);
      accQA = __builtin_amdgcn_wmma_f32_16x16x32_bf16(false, al.v, false, bha.v,
                                                      (short)0, accQA, false, false);
      accQB = __builtin_amdgcn_wmma_f32_16x16x32_bf16(false, al.v, false, bhb.v,
                                                      (short)0, accQB, false, false);
    }

    // C/D layout: lane holds column n=m; VGPR r -> row (8h + r).
    const float e2A = e2half[colA];
    const float e2B = e2half[colB];
    #pragma unroll
    for (int r = 0; r < 8; ++r) {
      float dA = e2A - (accPA[r] + accQA[r]);
      float dB = e2B - (accPB[r] + accQB[r]);
      if (dA < minv[r]) { minv[r] = dA; mini[r] = colA; }  // strict '<': first idx
      if (dB < minv[r]) { minv[r] = dB; mini[r] = colB; }
    }
  }

  // Cross-lane min+argmin over the 16 lanes of each half (rows 8h..8h+7).
  #pragma unroll
  for (int r = 0; r < 8; ++r) {
    float v = minv[r]; int i = mini[r];
    #pragma unroll
    for (int off = 1; off <= 8; off <<= 1) {
      float ov = __shfl_xor(v, off, 32);
      int   oi = __shfl_xor(i, off, 32);
      if (ov < v || (ov == v && oi < i)) { v = ov; i = oi; }
    }
    minv[r] = v; mini[r] = i;
  }
  if (m == 0) {
    #pragma unroll
    for (int r = 0; r < 8; ++r)
      out_idx[rowBase + 8 * h + r] = mini[r];   // stores codebook row (ref idx+1)
  }
}

// ---------------------------------------------------------------------------
// Per-row quantize + rotation trick + loss partial sums. One block per row.
// ---------------------------------------------------------------------------
__global__ void __launch_bounds__(128)
quantize_rotate(const float* __restrict__ z,
                const float* __restrict__ cb,
                const float* __restrict__ mask,
                const int* __restrict__ idx,
                float* __restrict__ zq,
                float* __restrict__ accum) {
  __shared__ float4 red4[128];
  const int n   = blockIdx.x;
  const int tid = threadIdx.x;
  const int t   = n & (T_ - 1);
  const int c0  = tid * 4;

  const float* zr = z + (size_t)n * C_;
  const int col   = idx[n];
  const float* qr = cb + (size_t)col * C_;

  float4 zv = *(const float4*)(zr + c0);
  float4 qv = *(const float4*)(qr + c0);
  float4 uv = make_float4(zv.x + qv.x, zv.y + qv.y, zv.z + qv.z, zv.w + qv.w);

  float th = zv.x * qv.x + zv.y * qv.y + zv.z * qv.z + zv.w * qv.w;
  float zz = zv.x * zv.x + zv.y * zv.y + zv.z * zv.z + zv.w * zv.w;
  float uu = uv.x * uv.x + uv.y * uv.y + uv.z * uv.z + uv.w * uv.w;
  float sd = 0.0f;
  if (t > 0) {
    float4 zp = *(const float4*)(zr - C_ + c0);
    float dx = zv.x - zp.x, dy = zv.y - zp.y, dz = zv.z - zp.z, dw = zv.w - zp.w;
    sd = dx * dx + dy * dy + dz * dz + dw * dw;
  }

  red4[tid] = make_float4(th, zz, uu, sd);
  __syncthreads();
  for (int o = 64; o > 0; o >>= 1) {
    if (tid < o) {
      float4 a = red4[tid], b = red4[tid + o];
      a.x += b.x; a.y += b.y; a.z += b.z; a.w += b.w;
      red4[tid] = a;
    }
    __syncthreads();
  }
  const float THv = red4[0].x, ZZv = red4[0].y, UUv = red4[0].z, SDv = red4[0].w;
  __syncthreads();

  const float mval  = mask[n];
  const float denom = fmaxf(sqrtf(UUv), 1e-12f);   // matches clamped L2 normalize
  const float cw    = 2.0f * (THv + ZZv) / (denom * denom);  // coeff on u=(z+q)
  const float cq    = 2.0f * ZZv;
  const bool  tm    = THv > THETA_THRESH;

  float4 ov;
  if (tm) {
    ov.x = (zv.x - cw * uv.x + cq * qv.x) * mval;
    ov.y = (zv.y - cw * uv.y + cq * qv.y) * mval;
    ov.z = (zv.z - cw * uv.z + cq * qv.z) * mval;
    ov.w = (zv.w - cw * uv.w + cq * qv.w) * mval;
  } else {                                          // straight-through forward = q
    ov.x = qv.x * mval; ov.y = qv.y * mval;
    ov.z = qv.z * mval; ov.w = qv.w * mval;
  }
  *(float4*)(zq + (size_t)n * C_ + c0) = ov;

  float dx = zv.x - ov.x, dy = zv.y - ov.y, dz = zv.z - ov.z, dw = zv.w - ov.w;
  float commit = dx * dx + dy * dy + dz * dz + dw * dw;

  red4[tid].x = commit;
  __syncthreads();
  for (int o = 64; o > 0; o >>= 1) {
    if (tid < o) red4[tid].x += red4[tid + o].x;
    __syncthreads();
  }
  if (tid == 0) {
    atomicAdd(accum + 0, red4[0].x * mval);  // commitment numerator
    atomicAdd(accum + 1, SDv * mval);        // smoothness numerator (mask[:,1:])
    atomicAdd(accum + 2, mval);              // mask sum
  }
}

__global__ void finalize(const float* __restrict__ accum, float* __restrict__ out) {
  const float valid = accum[2] * (float)C_;
  out[0] = accum[1] / valid;  // smoothness_loss
  out[1] = accum[0] / valid;  // commitment_loss
}

// ---------------------------------------------------------------------------
extern "C" void kernel_launch(void* const* d_in, const int* in_sizes, int n_in,
                              void* d_out, int out_size, void* d_ws, size_t ws_size,
                              hipStream_t stream) {
  (void)in_sizes; (void)n_in; (void)out_size; (void)ws_size;
  const float* z    = (const float*)d_in[0];
  const float* cb   = (const float*)d_in[1];
  const float* mask = (const float*)d_in[2];
  float* out = (float*)d_out;
  float* zq  = out + 2;

  char* ws = (char*)d_ws;
  unsigned short* cb_hi  = (unsigned short*)(ws);                        // 2 MB
  unsigned short* cb_lo  = (unsigned short*)(ws + (size_t)2 * 1024 * 1024);
  float*          e2half = (float*)(ws + (size_t)4 * 1024 * 1024);       // 8 KB
  int*            idxbuf = (int*)  (ws + (size_t)4 * 1024 * 1024 + 16 * 1024);
  float*          accum  = (float*)(ws + (size_t)4 * 1024 * 1024 + 96 * 1024);

  zero_accum<<<1, 32, 0, stream>>>(accum);
  prep_codebook<<<V_, 128, 0, stream>>>(cb, cb_hi, cb_lo, e2half);
  argmin_wmma<<<N_ / 16, 32, 0, stream>>>(z, cb_hi, cb_lo, e2half, idxbuf);
  quantize_rotate<<<N_, 128, 0, stream>>>(z, cb, mask, idxbuf, zq, accum);
  finalize<<<1, 1, 0, stream>>>(accum, out);
}
